// BiMambaBlock_79791902425748
// MI455X (gfx1250) — matrix operand order repaired and checked by reading, not compiled, not run on hardware
//
#include <hip/hip_runtime.h>
#include <hip/hip_bf16.h>

// ---------------- problem constants ----------------
#define DM   256          // d_model
#define DS   16           // d_state
#define DI   512          // d_inner
#define DTR  16           // dt_rank
#define DCONV 4
#define BB   2
#define LL   4096
#define MROWS (BB*LL)     // 8192 token rows
#define PROJW 64          // proj row width padded 48 -> 64

typedef __attribute__((ext_vector_type(16))) __bf16 v16bf;
typedef __attribute__((ext_vector_type(8)))  float  v8f;

__device__ __forceinline__ float silu_f(float x)     { return x / (1.f + __expf(-x)); }
__device__ __forceinline__ float softplus_f(float x) { return (x > 20.f) ? x : log1pf(__expf(x)); }
__device__ __forceinline__ float gelu_f(float x)     { return 0.5f * x * (1.f + erff(x * 0.70710678118654752f)); }

// ---------------- prep kernels ----------------
// Swizzle W[Ks,Ns] (f32 row-major) into WMMA B-fragment layout, zero-padded to [Kp,Np]:
// linear index = ((kblk*(Np/16) + ntile)*32 + lane)*16 + e
// where kg = kblk*32 + (lane>>4)*16 + e,  n = ntile*16 + (lane&15)
__global__ void swizzle_b_kernel(const float* __restrict__ W, __bf16* __restrict__ out,
                                 int Ks, int Ns, int Kp, int Np)
{
    const int tn = Np >> 4;
    const long total = (long)Kp * Np;
    for (long i = (long)blockIdx.x * blockDim.x + threadIdx.x; i < total;
         i += (long)gridDim.x * blockDim.x) {
        const int  e     = (int)(i & 15);
        const int  lane  = (int)((i >> 4) & 31);
        const long blk   = i >> 9;                 // kblk*tn + ntile
        const int  ntile = (int)(blk % tn);
        const int  kblk  = (int)(blk / tn);
        const int  kg = kblk * 32 + (lane >> 4) * 16 + e;
        const int  n  = ntile * 16 + (lane & 15);
        const float v = (kg < Ks && n < Ns) ? W[(long)kg * Ns + n] : 0.f;
        out[i] = (__bf16)v;
    }
}

__global__ void neg_exp_kernel(const float* __restrict__ s, float* __restrict__ d, int n) {
    int i = blockIdx.x * blockDim.x + threadIdx.x;
    if (i < n) d[i] = -__expf(s[i]);
}

// ---------------- fragment loaders ----------------
struct BFrag4 { v16bf b0, b1, b2, b3; };

__device__ __forceinline__ v16bf load_a_frag(const float* __restrict__ Arow, int k0, int half)
{
    // Contiguous k order (matches B swizzle): each lane holds one contiguous 16-float run.
    const float4 c0 = *(const float4*)(Arow + k0 + half * 16);
    const float4 c1 = *(const float4*)(Arow + k0 + half * 16 + 4);
    const float4 c2 = *(const float4*)(Arow + k0 + half * 16 + 8);
    const float4 c3 = *(const float4*)(Arow + k0 + half * 16 + 12);
    v16bf a;
    a[0]=(__bf16)c0.x;  a[1]=(__bf16)c0.y;  a[2]=(__bf16)c0.z;  a[3]=(__bf16)c0.w;
    a[4]=(__bf16)c1.x;  a[5]=(__bf16)c1.y;  a[6]=(__bf16)c1.z;  a[7]=(__bf16)c1.w;
    a[8]=(__bf16)c2.x;  a[9]=(__bf16)c2.y;  a[10]=(__bf16)c2.z; a[11]=(__bf16)c2.w;
    a[12]=(__bf16)c3.x; a[13]=(__bf16)c3.y; a[14]=(__bf16)c3.z; a[15]=(__bf16)c3.w;
    return a;
}

__device__ __forceinline__ BFrag4 load_b_frag4(const __bf16* __restrict__ bp)
{
    BFrag4 r;
    r.b0 = *(const v16bf*)(bp);
    r.b1 = *(const v16bf*)(bp + 512);
    r.b2 = *(const v16bf*)(bp + 1024);
    r.b3 = *(const v16bf*)(bp + 1536);
    return r;
}

__device__ __forceinline__ void wmma4(const v16bf& a, const BFrag4& b, v8f acc[4])
{
    acc[0] = __builtin_amdgcn_wmma_f32_16x16x32_bf16(false, a, false, b.b0, (short)0, acc[0], false, false);
    acc[1] = __builtin_amdgcn_wmma_f32_16x16x32_bf16(false, a, false, b.b1, (short)0, acc[1], false, false);
    acc[2] = __builtin_amdgcn_wmma_f32_16x16x32_bf16(false, a, false, b.b2, (short)0, acc[2], false, false);
    acc[3] = __builtin_amdgcn_wmma_f32_16x16x32_bf16(false, a, false, b.b3, (short)0, acc[3], false, false);
}

// ---------------- WMMA bf16 GEMM: 1 wave = 16x64 output, double-buffered K-loop ----------------
// C[M,N] = epilogue( A_f32[M,K](row stride lda) * B_swizzled )  (+ Cin)
// act: 0 none, 1 softplus(v+bias[n]), 2 gelu(v+bias[n]), 3 v+bias[n]
// Requires: K % 32 == 0, N % 64 == 0, M % 16 == 0, A rows 16B-aligned (lda % 4 == 0).
// launch_bounds(256, 4): cap at 4 waves/EU -> 256 VGPRs/wave, no spills for the
// double-buffered operand set (2x(8+32) + 32 acc + temps), still 4-way latency hiding.
__global__ void __launch_bounds__(256, 4)
gemm_wmma_bf16(const float* __restrict__ A, int lda,
               const __bf16* __restrict__ Bws,
               const float* __restrict__ bias,
               const float* __restrict__ Cin,
               float* __restrict__ C,
               int M, int N, int K, int act)
{
    const int lane = threadIdx.x & 31;
    const int wave = threadIdx.x >> 5;
    const int tn4  = N >> 6;                                  // 64-col super-tiles
    const long tiles = (long)(M >> 4) * tn4;
    const long tile  = (long)blockIdx.x * (blockDim.x >> 5) + wave;
    if (tile >= tiles) return;                                // wave-uniform
    const int tm   = (int)(tile / tn4);
    const int tc4  = (int)(tile % tn4);
    const int row0 = tm << 4, col0 = tc4 << 6;
    const int m    = lane & 15;
    const int half = lane >> 4;
    const int tn   = N >> 4;

    const float*  Arow = A + (long)(row0 + m) * lda;
    const __bf16* Bcol = Bws + (long)(col0 >> 4) * 512 + lane * 16;
    const long    bstride = (long)tn * 512;                   // per 32-K block

    v8f acc[4] = {{}, {}, {}, {}};

    const int kb = K >> 5;                                    // number of 32-wide K blocks
    // buffer 0 holds block i, buffer 1 holds block i+1 (unroll-by-2: no ping-pong copies)
    v16bf  a0 = load_a_frag(Arow, 0, half);
    BFrag4 b0 = load_b_frag4(Bcol);

    for (int i = 0; i < kb; i += 2) {
        if (i + 1 < kb) {                                     // uniform
            __builtin_prefetch(Arow + (i + 2) * 32, 0, 1);    // gfx1250 global_prefetch_b8
            v16bf  a1 = load_a_frag(Arow, (i + 1) * 32, half);
            BFrag4 b1 = load_b_frag4(Bcol + (long)(i + 1) * bstride);
            wmma4(a0, b0, acc);
            if (i + 2 < kb) {                                 // refill buffer 0 during b1's WMMAs
                __builtin_prefetch(Arow + (i + 3) * 32, 0, 1);
                a0 = load_a_frag(Arow, (i + 2) * 32, half);
                b0 = load_b_frag4(Bcol + (long)(i + 2) * bstride);
            }
            wmma4(a1, b1, acc);
        } else {
            wmma4(a0, b0, acc);                               // odd tail (e.g. K == 32)
        }
    }

#pragma unroll
    for (int t = 0; t < 4; ++t) {
        const int n = col0 + t * 16 + (lane & 15);
        const float bn = bias ? bias[n] : 0.f;
#pragma unroll
        for (int i = 0; i < 8; ++i) {
            const long idx = (long)(row0 + i + half * 8) * N + n;
            float v = acc[t][i];
            if (Cin) v += Cin[idx];
            if      (act == 1) v = softplus_f(v + bn);
            else if (act == 2) v = gelu_f(v + bn);
            else if (act == 3) v = v + bn;
            C[idx] = v;
        }
    }
}

// ---------------- causal depthwise conv + SiLU (direction-aware) ----------------
__global__ void conv_silu_kernel(const float* __restrict__ xz, const float* __restrict__ w,
                                 const float* __restrict__ bconv, float* __restrict__ xc, int rev)
{
    const long i = (long)blockIdx.x * blockDim.x + threadIdx.x;
    const long total = (long)BB * LL * DI;
    if (i >= total) return;
    const int  d  = (int)(i & (DI - 1));
    const long bl = i >> 9;
    const int  l  = (int)(bl & (LL - 1));
    const long b  = bl >> 12;
    float s = bconv[d];
#pragma unroll
    for (int k = 0; k < DCONV; ++k) {
        const int t = rev ? (l + (DCONV - 1) - k) : (l - (DCONV - 1) + k);
        if (t >= 0 && t < LL)
            s += xz[(b * LL + t) * (2 * DI) + d] * w[d * DCONV + k];
    }
    xc[i] = silu_f(s);
}

// ---------------- selective scan: 1 thread per (batch, channel), 16 states in regs ----------------
__global__ void scan_kernel(const float* __restrict__ delta, const float* __restrict__ xc,
                            const float* __restrict__ proj, const float* __restrict__ Aneg,
                            float* __restrict__ ys, int rev)
{
    const int idx = blockIdx.x * blockDim.x + threadIdx.x;
    if (idx >= BB * DI) return;
    const int d = idx & (DI - 1);
    const int b = idx >> 9;
    float Ad[DS], h[DS];
#pragma unroll
    for (int n = 0; n < DS; ++n) { Ad[n] = Aneg[d * DS + n]; h[n] = 0.f; }
    for (int step = 0; step < LL; ++step) {
        const int  t   = rev ? (LL - 1 - step) : step;
        const long pos = (long)b * LL + t;
        const float dl = delta[pos * DI + d];
        const float xv = xc[pos * DI + d];
        const float dx = dl * xv;
        const float* pr = proj + pos * PROJW;
        float y = 0.f;
#pragma unroll
        for (int n = 0; n < DS; ++n) {
            h[n] = __expf(dl * Ad[n]) * h[n] + dx * pr[DTR + n];   // B columns
            y   += h[n] * pr[DTR + DS + n];                        // C columns
        }
        ys[pos * DI + d] = y;
    }
}

// ---------------- skip + gate: y = (ys + xc*Dp) * silu(z) ----------------
__global__ void gate_kernel(float* __restrict__ ys, const float* __restrict__ xc,
                            const float* __restrict__ xz, const float* __restrict__ Dp)
{
    const long i = (long)blockIdx.x * blockDim.x + threadIdx.x;
    const long total = (long)BB * LL * DI;
    if (i >= total) return;
    const int  d  = (int)(i & (DI - 1));
    const long bl = i >> 9;
    const float z = xz[bl * (2 * DI) + DI + d];
    ys[i] = (ys[i] + xc[i] * Dp[d]) * silu_f(z);
}

// ---------------- LayerNorm(FF) * g2 + beta2 + residual Y ----------------
__global__ void ln_residual_kernel(const float* __restrict__ Y, const float* __restrict__ FF,
                                   const float* __restrict__ g2, const float* __restrict__ b2,
                                   float* __restrict__ out)
{
    __shared__ float red[DM];
    const long row = blockIdx.x;
    const int  c   = threadIdx.x;
    const float v  = FF[row * DM + c];
    red[c] = v;
    __syncthreads();
    for (int s = DM >> 1; s > 0; s >>= 1) { if (c < s) red[c] += red[c + s]; __syncthreads(); }
    const float mu = red[0] * (1.f / DM);
    __syncthreads();
    const float dv = v - mu;
    red[c] = dv * dv;
    __syncthreads();
    for (int s = DM >> 1; s > 0; s >>= 1) { if (c < s) red[c] += red[c + s]; __syncthreads(); }
    const float r = rsqrtf(red[0] * (1.f / DM) + 1e-5f);
    out[row * DM + c] = Y[row * DM + c] + dv * r * g2[c] + b2[c];
}

// ---------------- host orchestration ----------------
extern "C" void kernel_launch(void* const* d_in, const int* in_sizes, int n_in,
                              void* d_out, int out_size, void* d_ws, size_t ws_size,
                              hipStream_t stream)
{
    (void)in_sizes; (void)n_in; (void)out_size; (void)ws_size;
    const float* x       = (const float*)d_in[0];
    const float* W_in    = (const float*)d_in[1];
    const float* conv_w  = (const float*)d_in[2];
    const float* conv_b  = (const float*)d_in[3];
    const float* W_xproj = (const float*)d_in[4];
    const float* W_dt    = (const float*)d_in[5];
    const float* b_dt    = (const float*)d_in[6];
    const float* A_log   = (const float*)d_in[7];
    const float* Dp      = (const float*)d_in[8];
    const float* W_out   = (const float*)d_in[9];
    const float* W_ff1   = (const float*)d_in[10];
    const float* b_ff1   = (const float*)d_in[11];
    const float* W_ff2   = (const float*)d_in[12];
    const float* b_ff2   = (const float*)d_in[13];
    const float* g2      = (const float*)d_in[14];
    const float* beta2   = (const float*)d_in[15];
    float* out = (float*)d_out;
    char*  ws  = (char*)d_ws;

    size_t o = 0;
    auto alloc = [&](size_t bytes) { size_t r = o; o += (bytes + 255) & ~(size_t)255; return r; };
    const size_t o_win   = alloc((size_t)DM * 2 * DI * sizeof(__bf16));      // 256x1024
    const size_t o_wxp   = alloc((size_t)DI * PROJW * sizeof(__bf16));       // 512x64 (padded)
    const size_t o_wdt   = alloc((size_t)32 * DI * sizeof(__bf16));          // 32x512 (K padded)
    const size_t o_wout  = alloc((size_t)DI * DM * sizeof(__bf16));          // 512x256
    const size_t o_wff1  = alloc((size_t)DM * 4 * DM * sizeof(__bf16));      // 256x1024
    const size_t o_wff2  = alloc((size_t)4 * DM * DM * sizeof(__bf16));      // 1024x256
    const size_t o_Aneg  = alloc((size_t)DI * DS * sizeof(float));
    const size_t o_xz    = alloc((size_t)MROWS * 2 * DI * sizeof(float));    // 32 MB (reused as H)
    const size_t o_xc    = alloc((size_t)MROWS * DI * sizeof(float));
    const size_t o_proj  = alloc((size_t)MROWS * PROJW * sizeof(float));
    const size_t o_delta = alloc((size_t)MROWS * DI * sizeof(float));
    const size_t o_ys    = alloc((size_t)MROWS * DI * sizeof(float));
    const size_t o_Y     = alloc((size_t)MROWS * DM * sizeof(float));
    const size_t o_FF    = alloc((size_t)MROWS * DM * sizeof(float));

    __bf16* win_bf  = (__bf16*)(ws + o_win);
    __bf16* wxp_bf  = (__bf16*)(ws + o_wxp);
    __bf16* wdt_bf  = (__bf16*)(ws + o_wdt);
    __bf16* wout_bf = (__bf16*)(ws + o_wout);
    __bf16* wff1_bf = (__bf16*)(ws + o_wff1);
    __bf16* wff2_bf = (__bf16*)(ws + o_wff2);
    float* Aneg  = (float*)(ws + o_Aneg);
    float* xz    = (float*)(ws + o_xz);
    float* H     = xz;                       // FFN hidden reuses xz buffer
    float* xc    = (float*)(ws + o_xc);
    float* proj  = (float*)(ws + o_proj);
    float* delta = (float*)(ws + o_delta);
    float* ysb   = (float*)(ws + o_ys);
    float* Y     = (float*)(ws + o_Y);
    float* FF    = (float*)(ws + o_FF);

    auto swz = [&](const float* W, __bf16* dst, int Ks, int Ns, int Kp, int Np) {
        const long total = (long)Kp * Np;
        swizzle_b_kernel<<<(int)((total + 255) / 256), 256, 0, stream>>>(W, dst, Ks, Ns, Kp, Np);
    };
    swz(W_in,    win_bf,  DM,  2 * DI,       DM,  2 * DI);
    swz(W_xproj, wxp_bf,  DI,  DTR + 2 * DS, DI,  PROJW);    // N padded 48 -> 64
    swz(W_dt,    wdt_bf,  DTR, DI,           32,  DI);       // K padded 16 -> 32 (zero rows)
    swz(W_out,   wout_bf, DI,  DM,           DI,  DM);
    swz(W_ff1,   wff1_bf, DM,  4 * DM,       DM,  4 * DM);
    swz(W_ff2,   wff2_bf, 4 * DM, DM,        4 * DM, DM);
    neg_exp_kernel<<<(DI * DS + 255) / 256, 256, 0, stream>>>(A_log, Aneg, DI * DS);

    auto gemm = [&](const float* A, int lda, const __bf16* Bw, const float* bias,
                    const float* Cin, float* C, int M, int N, int K, int act) {
        const long tiles = (long)(M / 16) * (N / 64);
        const int  blocks = (int)((tiles + 7) / 8);          // 8 waves / block
        gemm_wmma_bf16<<<blocks, 256, 0, stream>>>(A, lda, Bw, bias, Cin, C, M, N, K, act);
    };

    const int nEl = BB * LL * DI;
    const int gEl = (nEl + 255) / 256;

    // xz = x @ W_in (shared by both directions: flip commutes with the matmul)
    gemm(x, DM, win_bf, nullptr, nullptr, xz, MROWS, 2 * DI, DM, 0);

    for (int dir = 0; dir < 2; ++dir) {
        conv_silu_kernel<<<gEl, 256, 0, stream>>>(xz, conv_w, conv_b, xc, dir);
        // proj (cols: dt 0..15, B 16..31, C 32..47, zero 48..63)
        gemm(xc, DI, wxp_bf, nullptr, nullptr, proj, MROWS, PROJW, DI, 0);
        // delta = softplus(dt @ W_dt + b_dt); K padded to 32 with zero weight rows
        gemm(proj, PROJW, wdt_bf, b_dt, nullptr, delta, MROWS, DI, 32, 1);
        scan_kernel<<<(BB * DI + 255) / 256, 256, 0, stream>>>(delta, xc, proj, Aneg, ysb, dir);
        gate_kernel<<<gEl, 256, 0, stream>>>(ysb, xc, xz, Dp);
        // Y (dir==0: write, dir==1: accumulate) = gated @ W_out
        gemm(ysb, DI, wout_bf, nullptr, dir ? Y : nullptr, Y, MROWS, DM, DI, 0);
    }

    // FFN: H = gelu(Y @ W_ff1 + b_ff1);  FF = H @ W_ff2 + b_ff2
    gemm(Y, DM, wff1_bf, b_ff1, nullptr, H, MROWS, 4 * DM, DM, 2);
    gemm(H, 4 * DM, wff2_bf, b_ff2, nullptr, FF, MROWS, DM, 4 * DM, 3);

    // out = Y + LayerNorm(FF) * g2 + beta2
    ln_residual_kernel<<<MROWS, 256, 0, stream>>>(Y, FF, g2, beta2, out);
}
